// HyenaFilter_49933289783824
// MI455X (gfx1250) — compile-verified, matrix operand order checked by reading, not run on hardware
//
#include <hip/hip_runtime.h>
#include <hip/hip_bf16.h>
#include <math.h>

// ---------------- Problem constants ----------------
#define BB   4
#define WCH  768
#define LSEQ 8192
#define EIN  33
#define ORD  64
#define NFFT 16384          // 2*LSEQ
#define KPAD 64             // E padded to 64 for K-multiple-of-32

typedef __attribute__((ext_vector_type(16))) _Float16 v16h;
typedef __attribute__((ext_vector_type(8)))  float    v8f;
typedef unsigned int u32x4 __attribute__((ext_vector_type(4)));
typedef int          i32x8 __attribute__((ext_vector_type(8)));
typedef int          i32x4 __attribute__((ext_vector_type(4)));

// ---------------------------------------------------------------------------
// WMMA fragment layouts per CDNA5 ISA 7.12.2 (wave32), f16 16x16x32:
//  A (16x32, MxK): lanes 0-15 -> row M=lane,   K chunks {0..7, 16..23}
//                  lanes 16-31 -> row M=lane-16, K chunks {8..15, 24..31}
//  B (32x16, KxN): lane = column N, same K chunking.
//  C/D (16x16 f32): VGPR r -> (M=r, N=lane) / (M=r+8, N=lane-16).
// Element e of the v16h maps to K = koff + (e<8 ? e : e+8), koff=(lane>>4)*8.
//
// Global operands are pre-swizzled into fragment-major order by the prep
// kernel: frag f, lane l occupy 16 contiguous halfs at ((f*32+l)*16).
// ---------------------------------------------------------------------------
__device__ inline v16h load_frag(const _Float16* __restrict__ p, int frag,
                                 int lane) {
  return *(const v16h*)(p + (((size_t)frag << 5) + lane) * 16);
}

// Gather-style A-fragment load (used for the small LDS activation tile).
__device__ inline v16h load_A_gather(const _Float16* base, int ld, int k0,
                                     int lane) {
  const int m    = lane & 15;
  const int koff = (lane >> 4) << 3;
  const _Float16* p = base + m * ld + k0 + koff;
  v16h a;
#pragma unroll
  for (int e = 0; e < 8; ++e) { a[e] = p[e]; a[8 + e] = p[16 + e]; }
  return a;
}

#define WMMA_F16(a, b, c)                                                    \
  __builtin_amdgcn_wmma_f32_16x16x32_f16(false, (a), false, (b), (short)0,   \
                                         (c), false, false)

// ---------------------------------------------------------------------------
// Tensor Data Mover: 1-row (contiguous) f32 tile, global -> LDS.
// D# per CDNA5 ISA: group0 = {count/type/addresses},
// group1 = {data_size, tensor/tile dims}, groups 2/3 unused (<=2D tensor).
// This toolchain exposes the 6-arg builtin (extra trailing i32x8 group).
// ---------------------------------------------------------------------------
__device__ inline void tdm_load_row_f32(const float* __restrict__ gsrc,
                                        void* lds_dst, int nelems) {
  const unsigned long long ga = (unsigned long long)(const void*)gsrc;
  const unsigned glo = (unsigned)ga, ghi = (unsigned)(ga >> 32);
  const unsigned lds = (unsigned)(unsigned long long)lds_dst; // LDS = addr[31:0]
  u32x4 g0;
  g0[0] = 1u;                                   // count=1, user mode, no gather
  g0[1] = lds;                                  // lds_addr [63:32]
  g0[2] = glo;                                  // global_addr [95:64]
  g0[3] = (ghi & 0x01FFFFFFu) | (2u << 30);     // global_addr hi | type=2
  i32x8 g1;
  g1[0] = (int)(2u << 16);                      // data_size=2 (4 bytes)
  g1[1] = (int)((unsigned)(nelems & 0xFFFF) << 16);   // tensor_dim0 lo16
  g1[2] = (int)(((unsigned)nelems >> 16) | (1u << 16)); // dim0 hi | tensor_dim1=1
  g1[3] = (int)((unsigned)(nelems & 0xFFFF) << 16);   // tile_dim0 [127:112]
  g1[4] = 1;                                    // tile_dim1=1
  g1[5] = nelems;                               // tensor_dim0_stride lo32
  g1[6] = (int)((unsigned)(nelems & 0xFFFF) << 16);   // tensor_dim1_stride lo16
  g1[7] = 0;
  const i32x4 gz4 = {0, 0, 0, 0};
  const i32x8 gz8 = {0, 0, 0, 0, 0, 0, 0, 0};
  __builtin_amdgcn_tensor_load_to_lds(g0, g1, gz4, gz4, gz8, 0);
}

// ---------------------------------------------------------------------------
// Kernel 1: pre-swizzle z and all weights into f16 WMMA-fragment order.
// ---------------------------------------------------------------------------
#define ZP_N   (LSEQ * KPAD)      // 1024 frags
#define W64_N  (KPAD * ORD)       // 8 frags per 64x64 matrix
#define WF_N   (ORD * WCH)        // 96 frags
__global__ void hyena_prep_f16(const float* __restrict__ z,
                               const float* __restrict__ w0,
                               const float* __restrict__ w1,
                               const float* __restrict__ w2,
                               const float* __restrict__ wf,
                               _Float16* __restrict__ zp,
                               _Float16* __restrict__ w0p,
                               _Float16* __restrict__ w1p,
                               _Float16* __restrict__ w2p,
                               _Float16* __restrict__ wfp) {
  int idx = blockIdx.x * blockDim.x + threadIdx.x;
  const int e    = idx & 15;
  const int lane = (idx >> 4) & 31;
  const int kin  = ((lane >> 4) << 3) + ((e < 8) ? e : e + 8); // K within 32
  if (idx < ZP_N) {                              // z A-fragments
    const int frag = idx >> 9, lt = frag >> 1, kt = frag & 1;
    const int m = lt * 16 + (lane & 15);
    const int k = kt * 32 + kin;
    zp[idx] = (_Float16)((k < EIN) ? z[m * EIN + k] : 0.0f);
    return;
  }
  idx -= ZP_N;
  if (idx < W64_N) {                             // w0 B-fragments (K padded)
    const int frag = idx >> 9, kt = frag >> 2, nt = frag & 3;
    const int n = nt * 16 + (lane & 15);
    const int k = kt * 32 + kin;
    w0p[idx] = (_Float16)((k < EIN) ? w0[k * ORD + n] : 0.0f);
    return;
  }
  idx -= W64_N;
  if (idx < W64_N) {                             // w1 B-fragments
    const int frag = idx >> 9, kt = frag >> 2, nt = frag & 3;
    const int n = nt * 16 + (lane & 15);
    const int k = kt * 32 + kin;
    w1p[idx] = (_Float16)w1[k * ORD + n];
    return;
  }
  idx -= W64_N;
  if (idx < W64_N) {                             // w2 B-fragments
    const int frag = idx >> 9, kt = frag >> 2, nt = frag & 3;
    const int n = nt * 16 + (lane & 15);
    const int k = kt * 32 + kin;
    w2p[idx] = (_Float16)w2[k * ORD + n];
    return;
  }
  idx -= W64_N;
  if (idx < WF_N) {                              // wf B-fragments
    const int frag = idx >> 9, kt = frag / 48, nt = frag % 48;
    const int n = nt * 16 + (lane & 15);
    const int k = kt * 32 + kin;
    wfp[idx] = (_Float16)wf[k * WCH + n];
    return;
  }
}

// ---------------------------------------------------------------------------
// Kernel 2: fused implicit-filter MLP via WMMA.  One block = 16 rows of L.
// ---------------------------------------------------------------------------
__global__ __launch_bounds__(256) void hyena_filter_mlp(
    const _Float16* __restrict__ zp,  const _Float16* __restrict__ w0p,
    const _Float16* __restrict__ w1p, const _Float16* __restrict__ w2p,
    const _Float16* __restrict__ wfp,
    const float* __restrict__ freq, const float* __restrict__ b0,
    const float* __restrict__ b1,   const float* __restrict__ b2,
    const float* __restrict__ deltas, float* __restrict__ kout) {
  __shared__ _Float16 hbuf[2][16 * ORD];
  const int tid  = threadIdx.x;
  const int lane = tid & 31;
  const int wave = tid >> 5;
  const int row0 = blockIdx.x * 16;

  const _Float16* const Wt[3] = {w0p, w1p, w2p};
  const float* const     Bs[3] = {b0, b1, b2};

#pragma unroll
  for (int layer = 0; layer < 3; ++layer) {
    const _Float16* Alds = &hbuf[(layer + 1) & 1][0];
    _Float16*       dst  = &hbuf[layer & 1][0];
    if (wave < 4) {                              // wave-uniform: EXEC all-1s
      const int nt = wave;
      v8f c = {};
      v16h a0 = (layer == 0) ? load_frag(zp, blockIdx.x * 2 + 0, lane)
                             : load_A_gather(Alds, ORD, 0, lane);
      v16h a1 = (layer == 0) ? load_frag(zp, blockIdx.x * 2 + 1, lane)
                             : load_A_gather(Alds, ORD, 32, lane);
      c = WMMA_F16(a0, load_frag(Wt[layer], 0 * 4 + nt, lane), c);
      c = WMMA_F16(a1, load_frag(Wt[layer], 1 * 4 + nt, lane), c);
      const int   n    = nt * 16 + (lane & 15);
      const float fq   = freq[n];
      const float bias = Bs[layer][n];
      const int   mb   = (lane >> 4) << 3;
#pragma unroll
      for (int r = 0; r < 8; ++r)
        dst[(mb + r) * ORD + n] = (_Float16)__sinf(fq * (c[r] + bias));
    }
    __syncthreads();
  }

  // Projection to 768 channels + exponential modulation; k stored [W][L].
  const _Float16* Alds = &hbuf[0][0];            // layer 2 wrote hbuf[0]
  const v16h pa0 = load_A_gather(Alds, ORD, 0, lane);
  const v16h pa1 = load_A_gather(Alds, ORD, 32, lane);
#pragma unroll
  for (int u = 0; u < 6; ++u) {
    const int nt = wave + u * 8;                 // 0..47
    v8f c = {};
    c = WMMA_F16(pa0, load_frag(wfp, 0 * 48 + nt, lane), c);
    c = WMMA_F16(pa1, load_frag(wfp, 1 * 48 + nt, lane), c);
    const int   n  = nt * 16 + (lane & 15);      // channel 0..767
    const float ad = fabsf(deltas[n]);
    const int   mb = (lane >> 4) << 3;
#pragma unroll
    for (int r = 0; r < 8; ++r) {
      const int   l = row0 + mb + r;
      const float t = (float)l * (1.0f / (float)(LSEQ - 1));
      kout[(size_t)n * LSEQ + l] = c[r] * __expf(-t * ad);
    }
  }
}

// ---------------------------------------------------------------------------
// In-place radix-2 FFT in LDS (16384 complex f32 = 128 KB).
// Forward: DIF, natural -> bit-reversed.  Inverse: DIT, bit-reversed ->
// natural.  Pointwise products are done in bit-reversed order.
// ---------------------------------------------------------------------------
__device__ inline void fft16k(float2* s, int tid, bool inverse) {
  if (!inverse) {
    for (int st = 13; st >= 0; --st) {
      const int span = 1 << st;
      for (int i = tid; i < (NFFT >> 1); i += 256) {
        const int j    = i & (span - 1);
        const int base = ((i >> st) << (st + 1)) + j;
        const float ang = -(float)M_PI * (float)j / (float)span;
        float sn, cs; __sincosf(ang, &sn, &cs);
        const float2 a = s[base], b = s[base + span];
        s[base]        = make_float2(a.x + b.x, a.y + b.y);
        const float dx = a.x - b.x, dy = a.y - b.y;
        s[base + span] = make_float2(dx * cs - dy * sn, dx * sn + dy * cs);
      }
      __syncthreads();
    }
  } else {
    for (int st = 0; st <= 13; ++st) {
      const int span = 1 << st;
      for (int i = tid; i < (NFFT >> 1); i += 256) {
        const int j    = i & (span - 1);
        const int base = ((i >> st) << (st + 1)) + j;
        const float ang = (float)M_PI * (float)j / (float)span;
        float sn, cs; __sincosf(ang, &sn, &cs);
        const float2 a = s[base], b = s[base + span];
        const float tx = b.x * cs - b.y * sn, ty = b.x * sn + b.y * cs;
        s[base]        = make_float2(a.x + tx, a.y + ty);
        s[base + span] = make_float2(a.x - tx, a.y - ty);
      }
      __syncthreads();
    }
  }
}

// Stage a contiguous f32 row into the upper half of the FFT buffer via the
// TDM (wave 0 issues, everyone barriers), then unpack to complex + zero-pad.
__device__ inline void stage_row_tdm(float2* s, const float* __restrict__ row,
                                     int tid) {
  float* stage = (float*)(s + LSEQ);             // 32 KB in upper half
  if (tid < 32) {                                // one wave issues the DMA
    tdm_load_row_f32(row, stage, LSEQ);
    __builtin_amdgcn_s_wait_tensorcnt(0);
  }
  __syncthreads();
  for (int i = tid; i < LSEQ; i += 256)
    s[i] = make_float2(stage[i], 0.0f);          // lower half only
  __syncthreads();
  for (int i = tid; i < LSEQ; i += 256)
    s[i + LSEQ] = make_float2(0.0f, 0.0f);       // now clobber staging
  __syncthreads();
}

// Kernel 3: per-channel filter spectrum kf[w] (bit-reversed order).
__global__ __launch_bounds__(256) void hyena_fft_filter(
    const float* __restrict__ k, float2* __restrict__ kf) {
  extern __shared__ float2 s[];
  const int w = blockIdx.x, tid = threadIdx.x;
  stage_row_tdm(s, k + (size_t)w * LSEQ, tid);
  fft16k(s, tid, false);
  float2* o = kf + (size_t)w * NFFT;
  for (int i = tid; i < NFFT; i += 256) o[i] = s[i];
}

// Kernel 4: per-(b,w) row: FFT(x) * kf, inverse FFT, +x*D.
__global__ __launch_bounds__(256) void hyena_fft_conv(
    const float* __restrict__ x, const float2* __restrict__ kf,
    const float* __restrict__ D, float* __restrict__ y) {
  extern __shared__ float2 s[];
  const int bid = blockIdx.x;                    // b*W + w
  const int w   = bid % WCH;
  const int tid = threadIdx.x;
  const float*  xr  = x + (size_t)bid * LSEQ;
  const float2* kfw = kf + (size_t)w * NFFT;

  stage_row_tdm(s, xr, tid);
  fft16k(s, tid, false);
  for (int i = tid; i < NFFT; i += 256) {        // product in bit-reversed order
    const float2 u = s[i], kv = kfw[i];
    s[i] = make_float2(u.x * kv.x - u.y * kv.y, u.x * kv.y + u.y * kv.x);
  }
  __syncthreads();
  fft16k(s, tid, true);
  const float dw = D[w];
  float* yr = y + (size_t)bid * LSEQ;
  for (int l = tid; l < LSEQ; l += 256)
    yr[l] = s[l].x * (1.0f / (float)NFFT) + xr[l] * dw;
}

// ---------------------------------------------------------------------------
extern "C" void kernel_launch(void* const* d_in, const int* in_sizes, int n_in,
                              void* d_out, int out_size, void* d_ws,
                              size_t ws_size, hipStream_t stream) {
  const float* x      = (const float*)d_in[0];
  const float* z      = (const float*)d_in[1];
  const float* freq   = (const float*)d_in[2];
  const float* w0     = (const float*)d_in[3];
  const float* b0     = (const float*)d_in[4];
  const float* w1     = (const float*)d_in[5];
  const float* b1     = (const float*)d_in[6];
  const float* w2     = (const float*)d_in[7];
  const float* b2     = (const float*)d_in[8];
  const float* wf     = (const float*)d_in[9];
  const float* deltas = (const float*)d_in[10];
  const float* Dp     = (const float*)d_in[11];
  float* out = (float*)d_out;

  char*  ws  = (char*)d_ws;
  size_t off = 0;
  auto carve = [&](size_t bytes) {
    size_t o = off;
    off = (off + bytes + 255) & ~(size_t)255;
    return o;
  };
  _Float16* zp  = (_Float16*)(ws + carve((size_t)ZP_N * 2));
  _Float16* w0p = (_Float16*)(ws + carve((size_t)W64_N * 2));
  _Float16* w1p = (_Float16*)(ws + carve((size_t)W64_N * 2));
  _Float16* w2p = (_Float16*)(ws + carve((size_t)W64_N * 2));
  _Float16* wfp = (_Float16*)(ws + carve((size_t)WF_N * 2));
  float*    kbf = (float*)(ws + carve((size_t)WCH * LSEQ * 4));
  float2*   kft = (float2*)(ws + carve((size_t)WCH * NFFT * 8));

  // 1) f16 conversion + fragment swizzle
  const int prep_n = ZP_N + 3 * W64_N + WF_N;
  hyena_prep_f16<<<(prep_n + 255) / 256, 256, 0, stream>>>(
      z, w0, w1, w2, wf, zp, w0p, w1p, w2p, wfp);

  // 2) WMMA MLP + modulation -> k[W,L]
  hyena_filter_mlp<<<LSEQ / 16, 256, 0, stream>>>(
      zp, w0p, w1p, w2p, wfp, freq, b0, b1, b2, deltas, kbf);

  // 3) filter spectra (128 KB dynamic LDS per block)
  (void)hipFuncSetAttribute(reinterpret_cast<const void*>(hyena_fft_filter),
                            hipFuncAttributeMaxDynamicSharedMemorySize,
                            NFFT * (int)sizeof(float2));
  (void)hipFuncSetAttribute(reinterpret_cast<const void*>(hyena_fft_conv),
                            hipFuncAttributeMaxDynamicSharedMemorySize,
                            NFFT * (int)sizeof(float2));
  hyena_fft_filter<<<WCH, 256, NFFT * sizeof(float2), stream>>>(kbf, kft);

  // 4) FFT convolution + residual
  hyena_fft_conv<<<BB * WCH, 256, NFFT * sizeof(float2), stream>>>(
      x, kft, Dp, out);
}
